// ExchangeBlock_31026843746428
// MI455X (gfx1250) — matrix operand order repaired and checked by reading, not compiled, not run on hardware
//
#include <hip/hip_runtime.h>
#include <hip/hip_bf16.h>

typedef __bf16 bf16_t;
typedef __attribute__((ext_vector_type(16))) __bf16 v16bf;
typedef __attribute__((ext_vector_type(8)))  float  v8f;

union B16U { v16bf v; uint4 u[2]; };

// ---- workspace layout (bf16 elements) ----
#define WS_WCAT 0        // 1344 x 32  -> 43008
#define WS_RW1  43008    //   64 x 64  ->  4096
#define WS_RW2  47104    //   64 x 64  ->  4096
#define WS_MW1  51200    //  128 x 64  ->  8192 (K padded 99->128 with zeros)
#define WS_MW2  59392    //   64 x 64  ->  4096
#define WS_TOTAL 63488

// B-operand swizzle: element (K,n) of a (Krows x 16*NT) matrix goes to
// tile (K/32)*NT + n/16 ; within tile lane = ((K>>3)&1)*16 + (n&15), elem = (K&7) + ((K>>4)&1)*8
__host__ __device__ inline int bswz(int K, int n, int NT) {
  return (((K >> 5) * NT + (n >> 4)) * 32 + (((K >> 3) & 1) * 16 + (n & 15))) * 16 + ((K & 7) + (((K >> 4) & 1) << 3));
}

__device__ inline v8f vzero() {
  v8f z = {0.f, 0.f, 0.f, 0.f, 0.f, 0.f, 0.f, 0.f};
  return z;
}

__device__ inline float silu_f(float x) { return x / (1.f + __expf(-x)); }

// K-slot owned by element e of this lane's A fragment (16-bit A layout, 16x32)
__device__ inline int kmap(int e, int half) {
  return half * 8 + (e & 7) + ((e >> 3) << 4);
}

// A operand from LDS row-major bf16 (for activation GEMMs): two ds_load_b128
__device__ inline v16bf load_a_tile(const bf16_t* s, int ld, int k0, int lane) {
  int m = lane & 15, half = (lane >> 4) & 1;
  const bf16_t* p = s + m * ld + k0 + half * 8;
  B16U t;
  t.u[0] = *reinterpret_cast<const uint4*>(p);
  t.u[1] = *reinterpret_cast<const uint4*>(p + 16);
  return t.v;
}

// B operand: pre-swizzled in global; lane's 16 elems contiguous (2x b128)
__device__ inline v16bf load_b_tile(const bf16_t* g, int tile, int lane) {
  const bf16_t* p = g + (size_t)(tile * 32 + lane) * 16;
  B16U t;
  t.u[0] = reinterpret_cast<const uint4*>(p)[0];
  t.u[1] = reinterpret_cast<const uint4*>(p)[1];
  return t.v;
}

__device__ inline v8f wmma_bf(v16bf a, v16bf b, v8f c) {
  return __builtin_amdgcn_wmma_f32_16x16x32_bf16(false, a, false, b, (short)0, c,
                                                 false, false);
}

// D/C layout: lane L holds col n=L&15, rows m = r + 8*(L>>4), r=0..7
__device__ inline void store_d_f32(float* s, int ld, int n0, int lane, v8f d) {
  int n = n0 + (lane & 15), half = lane >> 4;
#pragma unroll
  for (int r = 0; r < 8; ++r) s[(half * 8 + r) * ld + n] = d[r];
}

// ---------------- prep: convert + pre-swizzle weights to bf16 ----------------
__global__ void prep_kernel(const float* __restrict__ W0, const float* __restrict__ W1,
                            const float* __restrict__ W2, const float* __restrict__ rw1,
                            const float* __restrict__ rw2, const float* __restrict__ mw1,
                            const float* __restrict__ mw2, bf16_t* __restrict__ ws) {
  int i = blockIdx.x * blockDim.x + threadIdx.x;
  const float s0 = 0.0272772362f;            // 1/sqrt(1344)
  const float s1 = s0 * 0.57735027f;         // /sqrt(3)
  const float s2 = s0 * 0.44721360f;         // /sqrt(5)
  if (i < 43008) {                           // Wcat (1344 x 32), NT=2
    int K = i >> 5, n = i & 31;
    float val;
    if (K < 1024) {
      int u = K >> 5, v = K & 31;
      val = W0[(u * 32 + v) * 32 + n] * s0;
    } else if (K < 1280) {
      int r = K - 1024, u = r >> 4, v = r & 15;
      val = W1[(u * 16 + v) * 32 + n] * s1;
    } else {
      int r = K - 1280, u = r >> 3, v = r & 7;
      val = W2[(u * 8 + v) * 32 + n] * s2;
    }
    ws[WS_WCAT + bswz(K, n, 2)] = (bf16_t)val;
  } else if (i < 47104) {                    // rad_w1 (64 x 64), NT=4
    int j = i - 43008, K = j >> 6, n = j & 63;
    ws[WS_RW1 + bswz(K, n, 4)] = (bf16_t)rw1[K * 64 + n];
  } else if (i < 51200) {                    // rad_w2
    int j = i - 47104, K = j >> 6, n = j & 63;
    ws[WS_RW2 + bswz(K, n, 4)] = (bf16_t)rw2[K * 64 + n];
  } else if (i < 59392) {                    // mlp_w1 padded 99 -> 128 rows
    int j = i - 51200, K = j >> 6, n = j & 63;
    float v = (K < 99) ? mw1[K * 64 + n] : 0.f;
    ws[WS_MW1 + bswz(K, n, 4)] = (bf16_t)v;
  } else if (i < 63488) {                    // mlp_w2
    int j = i - 59392, K = j >> 6, n = j & 63;
    ws[WS_MW2 + bswz(K, n, 4)] = (bf16_t)mw2[K * 64 + n];
  }
}

// ---------------- main: one wave32 per 16-edge tile ----------------
__global__ __launch_bounds__(32) void edge_kernel(
    const float* __restrict__ nodes, const int* __restrict__ src,
    const int* __restrict__ dst, const int* __restrict__ batch_vec,
    const float* __restrict__ cell, const float* __restrict__ edge_shift,
    const float* __restrict__ pos, const float* __restrict__ angles,
    const float* __restrict__ cr_min, const float* __restrict__ cr_max,
    const float* __restrict__ ln_g, const float* __restrict__ ln_b,
    const float* __restrict__ rad_b1, const float* __restrict__ rad_b2,
    const float* __restrict__ mlp_b1, const float* __restrict__ mlp_b2,
    const float* __restrict__ out_w, const float* __restrict__ out_b,
    const bf16_t* __restrict__ ws, float* __restrict__ out, int E) {
  __shared__ __align__(16) float xs[16 * 120];
  __shared__ __align__(16) float xd[16 * 120];
  __shared__ __align__(16) float smix[16 * 32];
  __shared__ __align__(16) bf16_t sfeat[16 * 128];
  __shared__ __align__(16) bf16_t sh[16 * 64];
  __shared__ float sdist[16];
  __shared__ int ssrc[16], sdst[16];

  const int lane = threadIdx.x;
  const int e0 = blockIdx.x * 16;
  if (e0 >= E) return;
  const int m = lane & 15, half = lane >> 4;

  // ---- geometry + per-edge scalars (lanes 0..15) ----
  if (lane < 16) {
    int e = e0 + lane;
    int si = src[e], di = dst[e];
    ssrc[lane] = si;
    sdst[lane] = di;
    int b = batch_vec[si];
    const float* C = cell + b * 9;
    float s0 = edge_shift[e * 3 + 0], s1 = edge_shift[e * 3 + 1], s2 = edge_shift[e * 3 + 2];
    float tx = s0 * C[0] + s1 * C[3] + s2 * C[6];
    float ty = s0 * C[1] + s1 * C[4] + s2 * C[7];
    float tz = s0 * C[2] + s1 * C[5] + s2 * C[8];
    float rx = pos[di * 3 + 0] - pos[si * 3 + 0] + tx;
    float ry = pos[di * 3 + 1] - pos[si * 3 + 1] + ty;
    float rz = pos[di * 3 + 2] - pos[si * 3 + 2] + tz;
    sdist[lane] = sqrtf(rx * rx + ry * ry + rz * rz);
    // feat tail: angles / cr_min / cr_max + zero padding to K=128
    sfeat[lane * 128 + 96] = (bf16_t)angles[e];
    sfeat[lane * 128 + 97] = (bf16_t)cr_min[e];
    sfeat[lane * 128 + 98] = (bf16_t)cr_max[e];
    for (int c = 99; c < 128; ++c) sfeat[lane * 128 + c] = (bf16_t)0.f;
  }
  __syncthreads();

  // ---- gather node rows (16 x 120 f32, via b128) ----
  for (int i = lane; i < 16 * 30; i += 32) {
    int e = i / 30, q = i - e * 30;
    const float4* ps = reinterpret_cast<const float4*>(nodes + (size_t)ssrc[e] * 120);
    const float4* pd = reinterpret_cast<const float4*>(nodes + (size_t)sdst[e] * 120);
    reinterpret_cast<float4*>(xs)[i] = ps[q];
    reinterpret_cast<float4*>(xd)[i] = pd[q];
  }
  __syncthreads();

  // ---- tensor product: P(16x1344) @ Wcat(1344x32), A built in registers ----
  v8f accT0 = vzero(), accT1 = vzero();
  const float* xsr = xs + m * 120;   // this lane's A row (src features)
  const float* xdr = xd + m * 120;   // this lane's A row (dst features)

  // m0 region: chunks 0..31. A[m][K] = xs[m][kk] * xd[m][K]; xd side is
  // chunk-invariant per lane -> registers.
  {
    float axd[16];
#pragma unroll
    for (int e = 0; e < 16; ++e) axd[e] = xdr[kmap(e, half)];
    for (int kk = 0; kk < 32; ++kk) {
      float s = xsr[kk];
      v16bf a;
#pragma unroll
      for (int e = 0; e < 16; ++e) a[e] = (bf16_t)(axd[e] * s);
      accT0 = wmma_bf(a, load_b_tile(ws + WS_WCAT, kk * 2 + 0, lane), accT0);
      accT1 = wmma_bf(a, load_b_tile(ws + WS_WCAT, kk * 2 + 1, lane), accT1);
    }
  }
  // m1 region: chunks 32..39. K-slot e -> u = 2c + (e>>3), v = half*8 + (e&7).
  {
    float vsd[24];
#pragma unroll
    for (int j = 0; j < 8; ++j)
#pragma unroll
      for (int mm = 0; mm < 3; ++mm)
        vsd[j * 3 + mm] = xdr[32 + 3 * (half * 8 + j) + mm];
    for (int c = 0; c < 8; ++c) {
      float u0[3], u1[3];
#pragma unroll
      for (int mm = 0; mm < 3; ++mm) {
        u0[mm] = xsr[32 + 3 * (2 * c + 0) + mm];
        u1[mm] = xsr[32 + 3 * (2 * c + 1) + mm];
      }
      v16bf a;
#pragma unroll
      for (int e = 0; e < 16; ++e) {
        int j = e & 7;
        const float* uu = (e >> 3) ? u1 : u0;
        a[e] = (bf16_t)(uu[0] * vsd[j * 3 + 0] + uu[1] * vsd[j * 3 + 1] +
                        uu[2] * vsd[j * 3 + 2]);
      }
      int kk = 32 + c;
      accT0 = wmma_bf(a, load_b_tile(ws + WS_WCAT, kk * 2 + 0, lane), accT0);
      accT1 = wmma_bf(a, load_b_tile(ws + WS_WCAT, kk * 2 + 1, lane), accT1);
    }
  }
  // m2 region: chunks 40..41. K-slot e -> u = 4c + half + 2*(e>>3), v = e&7.
  {
    float tsd[40];
#pragma unroll
    for (int j = 0; j < 8; ++j)
#pragma unroll
      for (int mm = 0; mm < 5; ++mm)
        tsd[j * 5 + mm] = xdr[80 + 5 * j + mm];
    for (int c = 0; c < 2; ++c) {
      float ua[5], ub[5];
#pragma unroll
      for (int mm = 0; mm < 5; ++mm) {
        ua[mm] = xsr[80 + 5 * (c * 4 + half + 0) + mm];
        ub[mm] = xsr[80 + 5 * (c * 4 + half + 2) + mm];
      }
      v16bf a;
#pragma unroll
      for (int e = 0; e < 16; ++e) {
        int j = e & 7;
        const float* uu = (e >> 3) ? ub : ua;
        float s = 0.f;
#pragma unroll
        for (int mm = 0; mm < 5; ++mm) s += uu[mm] * tsd[j * 5 + mm];
        a[e] = (bf16_t)s;
      }
      int kk = 40 + c;
      accT0 = wmma_bf(a, load_b_tile(ws + WS_WCAT, kk * 2 + 0, lane), accT0);
      accT1 = wmma_bf(a, load_b_tile(ws + WS_WCAT, kk * 2 + 1, lane), accT1);
    }
  }

  // ---- layernorm over 32 channels -> sfeat[:, 0:32] ----
  store_d_f32(smix, 32, 0, lane, accT0);
  store_d_f32(smix, 32, 16, lane, accT1);
  __syncthreads();
  if (lane < 16) {
    float mu = 0.f;
#pragma unroll
    for (int c = 0; c < 32; ++c) mu += smix[lane * 32 + c];
    mu *= (1.f / 32.f);
    float var = 0.f;
#pragma unroll
    for (int c = 0; c < 32; ++c) {
      float d = smix[lane * 32 + c] - mu;
      var += d * d;
    }
    var *= (1.f / 32.f);
    float inv = rsqrtf(var + 1e-5f);
#pragma unroll
    for (int c = 0; c < 32; ++c)
      sfeat[lane * 128 + c] = (bf16_t)((smix[lane * 32 + c] - mu) * inv * ln_g[c] + ln_b[c]);
  }

  const int n = lane & 15;

  // ---- radial MLP layer 1: RBF A fragment built in registers ----
  {
    float distm = sdist[m];
    v8f acc[4] = {vzero(), vzero(), vzero(), vzero()};
    for (int kk = 0; kk < 2; ++kk) {
      v16bf a;
#pragma unroll
      for (int e = 0; e < 16; ++e) {
        int K = kk * 32 + kmap(e, half);
        float dd = distm - (float)K * (5.0f / 63.0f);
        a[e] = (bf16_t)__expf(-10.0f * dd * dd);
      }
#pragma unroll
      for (int j = 0; j < 4; ++j)
        acc[j] = wmma_bf(a, load_b_tile(ws + WS_RW1, kk * 4 + j, lane), acc[j]);
    }
#pragma unroll
    for (int j = 0; j < 4; ++j) {
      float bb = rad_b1[j * 16 + n];
#pragma unroll
      for (int r = 0; r < 8; ++r)
        sh[(half * 8 + r) * 64 + j * 16 + n] = (bf16_t)silu_f(acc[j][r] + bb);
    }
  }
  __syncthreads();

  // ---- radial MLP layer 2 -> sfeat[:, 32:96] ----
  {
    v8f acc[4] = {vzero(), vzero(), vzero(), vzero()};
    for (int kk = 0; kk < 2; ++kk) {
      v16bf a = load_a_tile(sh, 64, kk * 32, lane);
#pragma unroll
      for (int j = 0; j < 4; ++j)
        acc[j] = wmma_bf(a, load_b_tile(ws + WS_RW2, kk * 4 + j, lane), acc[j]);
    }
#pragma unroll
    for (int j = 0; j < 4; ++j) {
      float bb = rad_b2[j * 16 + n];
#pragma unroll
      for (int r = 0; r < 8; ++r)
        sfeat[(half * 8 + r) * 128 + 32 + j * 16 + n] = (bf16_t)(acc[j][r] + bb);
    }
  }
  __syncthreads();

  // ---- edge MLP layer 1: (16x128)@(128x64) + silu -> sh ----
  {
    v8f acc[4] = {vzero(), vzero(), vzero(), vzero()};
    for (int kk = 0; kk < 4; ++kk) {
      v16bf a = load_a_tile(sfeat, 128, kk * 32, lane);
#pragma unroll
      for (int j = 0; j < 4; ++j)
        acc[j] = wmma_bf(a, load_b_tile(ws + WS_MW1, kk * 4 + j, lane), acc[j]);
    }
    __syncthreads();
#pragma unroll
    for (int j = 0; j < 4; ++j) {
      float bb = mlp_b1[j * 16 + n];
#pragma unroll
      for (int r = 0; r < 8; ++r)
        sh[(half * 8 + r) * 64 + j * 16 + n] = (bf16_t)silu_f(acc[j][r] + bb);
    }
  }
  __syncthreads();

  // ---- edge MLP layer 2 + silu -> hbuf (reuse xs as f32 16x64) ----
  float* hbuf = xs;
  {
    v8f acc[4] = {vzero(), vzero(), vzero(), vzero()};
    for (int kk = 0; kk < 2; ++kk) {
      v16bf a = load_a_tile(sh, 64, kk * 32, lane);
#pragma unroll
      for (int j = 0; j < 4; ++j)
        acc[j] = wmma_bf(a, load_b_tile(ws + WS_MW2, kk * 4 + j, lane), acc[j]);
    }
    __syncthreads();
#pragma unroll
    for (int j = 0; j < 4; ++j) {
      float bb = mlp_b2[j * 16 + n];
#pragma unroll
      for (int r = 0; r < 8; ++r)
        hbuf[(half * 8 + r) * 64 + j * 16 + n] = silu_f(acc[j][r] + bb);
    }
  }
  __syncthreads();

  // ---- output projection (64 -> 1) ----
  if (lane < 16) {
    float a = out_b[0];
#pragma unroll
    for (int c = 0; c < 64; ++c) a += hbuf[lane * 64 + c] * out_w[c];
    out[e0 + lane] = a;
  }
}

extern "C" void kernel_launch(void* const* d_in, const int* in_sizes, int n_in,
                              void* d_out, int out_size, void* d_ws, size_t ws_size,
                              hipStream_t stream) {
  (void)n_in; (void)out_size; (void)ws_size;
  const float* nodes      = (const float*)d_in[0];
  const int*   src        = (const int*)d_in[1];
  const int*   dst        = (const int*)d_in[2];
  const int*   batch_vec  = (const int*)d_in[3];
  const float* cell       = (const float*)d_in[4];
  const float* edge_shift = (const float*)d_in[5];
  const float* pos        = (const float*)d_in[6];
  const float* angles     = (const float*)d_in[7];
  const float* cr_mn      = (const float*)d_in[8];
  const float* cr_mx      = (const float*)d_in[9];
  const float* W0         = (const float*)d_in[10];
  const float* W1         = (const float*)d_in[11];
  const float* W2         = (const float*)d_in[12];
  const float* ln_g       = (const float*)d_in[13];
  const float* ln_b       = (const float*)d_in[14];
  const float* rw1        = (const float*)d_in[15];
  const float* rb1        = (const float*)d_in[16];
  const float* rw2        = (const float*)d_in[17];
  const float* rb2        = (const float*)d_in[18];
  const float* mw1        = (const float*)d_in[19];
  const float* mb1        = (const float*)d_in[20];
  const float* mw2        = (const float*)d_in[21];
  const float* mb2        = (const float*)d_in[22];
  const float* ow         = (const float*)d_in[23];
  const float* ob         = (const float*)d_in[24];
  bf16_t* ws = (bf16_t*)d_ws;
  int E = in_sizes[1];

  prep_kernel<<<(WS_TOTAL + 255) / 256, 256, 0, stream>>>(W0, W1, W2, rw1, rw2, mw1, mw2, ws);
  edge_kernel<<<(E + 15) / 16, 32, 0, stream>>>(
      nodes, src, dst, batch_vec, cell, edge_shift, pos, angles, cr_mn, cr_mx,
      ln_g, ln_b, rb1, rb2, mb1, mb2, ow, ob, ws, (float*)d_out, E);
}
